// HungarianLoss_10058813407503
// MI455X (gfx1250) — matrix operand order repaired and checked by reading, not compiled, not run on hardware
//
#include <hip/hip_runtime.h>

// Problem shape (fixed by the reference).
#define BATCH 128
#define NQ    192      // queries == targets per sample (square assignment)
#define NC    512      // classes
#define BIGF  1e30f

typedef __attribute__((ext_vector_type(16))) _Float16 v16h;
typedef __attribute__((ext_vector_type(8)))  _Float16 v8h;
typedef __attribute__((ext_vector_type(8)))  float    v8f;

// ---- dynamic shared memory layout (bytes) ----
// cost  : 192*192 f32 = 147456
// probS : 16*512 f16  = 16384   (one 16-row tile of softmax probs, f16)
// small arrays of 192 entries each after that.
#define OFF_COST   0
#define OFF_PROB   147456
#define OFF_RMAX   163840
#define OFF_RSUM   164608
#define OFF_U      165376
#define OFF_SPC    166144
#define OFF_NLL    166912
#define OFF_TGT    167680
#define OFF_PATH   168448
#define OFF_R4C    169216
#define OFF_C4R    169984
#define OFF_SR     170752
#define SMEM_BYTES 171520

extern "C" __global__ __launch_bounds__(384)
void hungarian_loss_kernel(const float* __restrict__ pred,
                           const int*  __restrict__ tgt,
                           float* __restrict__ ws)
{
    extern __shared__ __align__(16) char smem[];
    float*    costS = (float*)   (smem + OFF_COST);
    _Float16* probS = (_Float16*)(smem + OFF_PROB);
    float*    rmaxS = (float*)   (smem + OFF_RMAX);
    float*    rsumS = (float*)   (smem + OFF_RSUM);
    float*    uS    = (float*)   (smem + OFF_U);
    float*    spcS  = (float*)   (smem + OFF_SPC);
    float*    nllS  = (float*)   (smem + OFF_NLL);
    int*      tgtS  = (int*)     (smem + OFF_TGT);
    int*      pathS = (int*)     (smem + OFF_PATH);
    int*      r4cS  = (int*)     (smem + OFF_R4C);   // row assigned to column j
    int*      c4rS  = (int*)     (smem + OFF_C4R);   // column assigned to row i
    int*      srS   = (int*)     (smem + OFF_SR);

    const int b    = blockIdx.x;
    const int tid  = threadIdx.x;       // 0..383  (12 wave32)
    const int wid  = tid >> 5;          // 0..11
    const int lane = tid & 31;
    const size_t base = (size_t)b * NQ * NC;

    // ---------------- Phase 0: targets -> LDS ----------------
    for (int j = tid; j < NQ; j += 384) tgtS[j] = tgt[b * NQ + j];

    // ---------------- Phase 1: per-row softmax stats ----------------
    // One wave per row, 16 columns per lane (512 = 32*16).
    for (int r = wid; r < NQ; r += 12) {
        const float* row = pred + base + (size_t)r * NC;
        float vals[16];
        float mx = -BIGF;
#pragma unroll
        for (int k = 0; k < 16; ++k) { vals[k] = row[lane + 32 * k]; mx = fmaxf(mx, vals[k]); }
#pragma unroll
        for (int off = 16; off >= 1; off >>= 1) mx = fmaxf(mx, __shfl_xor(mx, off, 32));
        float s = 0.f;
#pragma unroll
        for (int k = 0; k < 16; ++k) s += __expf(vals[k] - mx);
#pragma unroll
        for (int off = 16; off >= 1; off >>= 1) s += __shfl_xor(s, off, 32);
        if (lane == 0) { rmaxS[r] = mx; rsumS[r] = s; }
    }
    __syncthreads();

    // ---------------- Phase 2: cost = -(P_f16 @ OneHot(tgt)^T) via WMMA ----------------
    // 12 row-tiles of 16 queries; each of the 12 waves owns one 16-wide column tile.
    const int l15 = lane & 15;
    const int hi  = lane >> 4;          // 0: lanes 0-15, 1: lanes 16-31
    const int jt  = wid;                // column tile for this wave
    const int jcol = jt * 16 + l15;
    const int tj  = tgtS[jcol];         // target class for B-fragment one-hot column

    for (int qt = 0; qt < 12; ++qt) {
        // stage 16 rows of f16 softmax probs into LDS
        for (int idx = tid; idx < 16 * NC; idx += 384) {
            int r = idx >> 9, c = idx & (NC - 1);
            int q = qt * 16 + r;
            float x = pred[base + (size_t)q * NC + c];
            float p = __expf(x - rmaxS[q]) / rsumS[q];
            probS[idx] = (_Float16)p;
        }
        __syncthreads();

        v8f acc = {};
        for (int c0 = 0; c0 < NC; c0 += 32) {
            // A fragment (16x32 f16): lane group layout per CDNA5 ISA 7.12.2:
            // elements 0..7  = K in [c0 + hi*8,      +8)
            // elements 8..15 = K in [c0 + 16 + hi*8, +8)
            const v8h a_lo = *(const v8h*)&probS[l15 * NC + c0 + hi * 8];
            const v8h a_hi = *(const v8h*)&probS[l15 * NC + c0 + 16 + hi * 8];
            v16h a;
#pragma unroll
            for (int k = 0; k < 8; ++k) { a[k] = a_lo[k]; a[k + 8] = a_hi[k]; }

            // B fragment (32x16 f16 one-hot): lane n holds column jcol,
            // element e corresponds to K = c0 + (e & 8)*2 + hi*8 + (e & 7).
            const int rel = tj - c0;
            v16h bf;
#pragma unroll
            for (int k = 0; k < 16; ++k) bf[k] = (_Float16)((rel == ((k & 8) << 1) + hi * 8 + (k & 7)) ? 1.0f : 0.0f);

            acc = __builtin_amdgcn_wmma_f32_16x16x32_f16(
                false, a, false, bf, (short)0, acc, false, false);
        }
        // C layout: VGPR r -> row (r + hi*8), col = l15 within the tile.
#pragma unroll
        for (int r = 0; r < 8; ++r) {
            int row = qt * 16 + r + hi * 8;
            costS[row * NQ + jt * 16 + l15] = -acc[r];
        }
        __syncthreads();
    }

    // ---------------- Phase 3: Jonker-Volgenant LSAP, single wave, no block barriers ----------------
    if (wid == 0) {
        float vloc[6];                          // dual v, 6 columns per lane (j = lane + 32t)
#pragma unroll
        for (int t = 0; t < 6; ++t) {
            int j = lane + 32 * t;
            vloc[t] = 0.f; uS[j] = 0.f; r4cS[j] = -1; c4rS[j] = -1;
        }
        __builtin_amdgcn_wave_barrier();

        for (int cur = 0; cur < NQ; ++cur) {
            float spcl[6]; bool scl[6];
#pragma unroll
            for (int t = 0; t < 6; ++t) { spcl[t] = BIGF; scl[t] = false; srS[lane + 32 * t] = 0; }
            int   i = cur, sink = -1;
            float minVal = 0.f;

            while (sink < 0) {
                if (lane == 0) srS[i] = 1;
                const float ui = uS[i];
                const float* crow = &costS[i * NQ];
                // relax the 6 columns owned by this lane
#pragma unroll
                for (int t = 0; t < 6; ++t) {
                    int j = lane + 32 * t;
                    float r = minVal + crow[j] - ui - vloc[t];
                    if (!scl[t] && r < spcl[t]) { spcl[t] = r; spcS[j] = r; pathS[j] = i; }
                }
                // masked argmin (first-min tie-break like jnp.argmin)
                float bv = BIGF; int bj = NQ;
#pragma unroll
                for (int t = 0; t < 6; ++t) {
                    float m = scl[t] ? BIGF : spcl[t];
                    int   j = lane + 32 * t;
                    if (m < bv || (m == bv && j < bj)) { bv = m; bj = j; }
                }
#pragma unroll
                for (int off = 16; off >= 1; off >>= 1) {
                    float ov = __shfl_xor(bv, off, 32);
                    int   oj = __shfl_xor(bj, off, 32);
                    if (ov < bv || (ov == bv && oj < bj)) { bv = ov; bj = oj; }
                }
                minVal = bv;
                const int j = bj;
                if ((j & 31) == lane) scl[j >> 5] = true;   // SC[j] = true (owner lane)
                const int r4 = r4cS[j];
                if (r4 < 0) sink = j; else i = r4;
            }

            // dual updates (pre-augmentation col4row, matching the reference)
            if (lane == 0) uS[cur] += minVal;
#pragma unroll
            for (int t = 0; t < 6; ++t) {
                int k = lane + 32 * t;
                if (srS[k] && k != cur) uS[k] += minVal - spcS[c4rS[k]];
            }
#pragma unroll
            for (int t = 0; t < 6; ++t)
                if (scl[t]) vloc[t] -= (minVal - spcl[t]);

            // augment along the stored path (serial, lane 0)
            if (lane == 0) {
                int j = sink;
                while (true) {
                    int ii = pathS[j];
                    r4cS[j] = ii;
                    int nj = c4rS[ii];
                    c4rS[ii] = j;
                    if (ii == cur) break;
                    j = nj;
                }
            }
            __builtin_amdgcn_wave_barrier();
        }
    }
    __syncthreads();

    // ---------------- Phase 4: cross-entropy of matched classes ----------------
    if (tid < NQ) {
        const int q   = tid;
        const int j   = c4rS[q];          // matched column for query q
        const int tc  = tgtS[j];          // target class
        const float x = pred[base + (size_t)q * NC + tc];
        nllS[q] = -(x - rmaxS[q] - __logf(rsumS[q]));
    }
    __syncthreads();
    if (tid == 0) {
        float s = 0.f;
        for (int q = 0; q < NQ; ++q) s += nllS[q];
        ws[b] = s;
    }
}

extern "C" __global__ void hungarian_loss_reduce(const float* __restrict__ ws,
                                                 float* __restrict__ out)
{
    if (threadIdx.x == 0 && blockIdx.x == 0) {
        double s = 0.0;
        for (int b = 0; b < BATCH; ++b) s += (double)ws[b];
        out[0] = (float)(s / (double)(BATCH * NQ));
    }
}

extern "C" void kernel_launch(void* const* d_in, const int* in_sizes, int n_in,
                              void* d_out, int out_size, void* d_ws, size_t ws_size,
                              hipStream_t stream)
{
    (void)in_sizes; (void)n_in; (void)out_size; (void)ws_size;
    const float* pred = (const float*)d_in[0];
    const int*   tgt  = (const int*)d_in[1];
    float* ws  = (float*)d_ws;       // 128 per-sample partial sums
    float* out = (float*)d_out;

    hungarian_loss_kernel<<<dim3(BATCH), dim3(384), SMEM_BYTES, stream>>>(pred, tgt, ws);
    hungarian_loss_reduce<<<dim3(1), dim3(32), 0, stream>>>(ws, out);
}
